// MelEmbeddingModelMelSmoothResidualUpsampling_35957466202498
// MI455X (gfx1250) — compile-verified, matrix operand order checked by reading, not run on hardware
//
#include <hip/hip_runtime.h>
#include <stdint.h>

// ---------------------------------------------------------------------------
// MelEmbedding model for MI455X (gfx1250): conv blocks (VALU), LSTM input
// projections + recurrence + post/up projections on v_wmma_f32_16x16x32_f16.
// Recurrence runs as one persistent 32-wave workgroup with ~249KB of LDS
// state; pre-activations are double-buffer staged into LDS with
// global_load_async_to_lds_b128 (ASYNCcnt) so the serial chain never waits
// on global memory.
// ---------------------------------------------------------------------------

typedef __attribute__((ext_vector_type(16))) _Float16 v16h;
typedef __attribute__((ext_vector_type(8)))  _Float16 v8h;
typedef __attribute__((ext_vector_type(8)))  float    v8f;

#define BB   64
#define SS   1024
#define MEL  60
#define HH   180
#define H4   720
#define POST 8192
#define OUTN 300
#define HPAD 192   // K-pad of H to multiple of 32
#define K0P  64    // K-pad of MEL for layer-0 input proj

// Fragment loader: 16 contiguous f16 K-values split 8+8 per the CDNA5 WMMA
// 16-bit A/B layout (lane%16 = row, lane/16 selects K-subgroup; VGPRs 0-3 hold
// K = 8*half+0..7, VGPRs 4-7 hold K = 16+8*half+0..7).
template <typename P>
__device__ __forceinline__ v16h frag_ld(P p) {
  v8h lo = *(const v8h*)(p);
  v8h hi = *(const v8h*)(p + 16);
  return __builtin_shufflevector(lo, hi, 0,1,2,3,4,5,6,7,8,9,10,11,12,13,14,15);
}

__device__ __forceinline__ float sigmoidf_(float x) {
  return 1.0f / (1.0f + __expf(-x));
}

// Low 32 bits of a generic pointer to LDS = wave-relative LDS byte offset
// (ISA 10.2).
__device__ __forceinline__ unsigned lds_off32(const void* p) {
  return (unsigned)(uintptr_t)p;
}

// ------------------------------ mel smooth blocks ---------------------------
// x layout (B, S, MEL). out[b,s,c] = src[b,s,c] + bias[j,i]
//   + sum_{ci,k} src[b, s+k-2, 3i+ci+d_j] * w[j,i,ci,k],  i=c/3, j=c%3,
//   d = {-1,0,+1}[j], zero-padded in both channel and time.
__global__ void __launch_bounds__(256) mel_block_kernel(
    const float* __restrict__ src, const float* __restrict__ wAll,
    const float* __restrict__ bAll, float* __restrict__ dst, int blk) {
  int idx = blockIdx.x * 256 + threadIdx.x;
  if (idx >= BB * SS * MEL) return;
  int c = idx % MEL;
  int s = (idx / MEL) % SS;
  int b = idx / (MEL * SS);
  int i = c / 3, j = c % 3;
  int d = j - 1;
  const float* w = wAll + (((size_t)blk * 3 + j) * 20 + i) * 15;
  float acc = bAll[((size_t)blk * 3 + j) * 20 + i] + src[idx];
  const float* sb = src + (size_t)b * SS * MEL;
#pragma unroll
  for (int ci = 0; ci < 3; ++ci) {
    int ch = 3 * i + ci + d;
    if (ch < 0 || ch >= MEL) continue;
#pragma unroll
    for (int k = 0; k < 5; ++k) {
      int ss = s + k - 2;
      if (ss < 0 || ss >= SS) continue;
      acc += sb[(size_t)ss * MEL + ch] * w[ci * 5 + k];
    }
  }
  dst[idx] = acc;
}

// ------------------------------ packing helpers -----------------------------
__global__ void __launch_bounds__(256) pack_a0_kernel(
    const float* __restrict__ src, _Float16* __restrict__ dst) {
  int idx = blockIdx.x * 256 + threadIdx.x;            // (B*S, K0P)
  if (idx >= BB * SS * K0P) return;
  int m = idx / K0P, k = idx % K0P;
  dst[idx] = (k < MEL) ? (_Float16)src[(size_t)m * MEL + k] : (_Float16)0.0f;
}

__global__ void __launch_bounds__(256) pack_w_kernel(
    const float* __restrict__ src, _Float16* __restrict__ dst,
    int rows, int srcRows, int K, int Kp) {
  int idx = blockIdx.x * 256 + threadIdx.x;
  if (idx >= rows * Kp) return;
  int r = idx / Kp, k = idx % Kp;
  float v = (r < srcRows && k < K) ? src[(size_t)r * K + k] : 0.0f;
  dst[idx] = (_Float16)v;
}

__global__ void __launch_bounds__(256) bias_comb_kernel(
    const float* __restrict__ bih, const float* __restrict__ bhh,
    float* __restrict__ comb) {
  int idx = blockIdx.x * 256 + threadIdx.x;
  if (idx < 4 * H4) comb[idx] = bih[idx] + bhh[idx];
}

// --------------------- LSTM input projection (big GEMM) ---------------------
// pre[t,b,n] = sum_k A[b*S+t, k] * W[n, k] + bias[n];  f16 WMMA, f32 acc.
// One wave per 16x16 tile; M = 65536 (4096 tiles), N = 720 (45 tiles).
__global__ void __launch_bounds__(256) gemm_pre_kernel(
    const _Float16* __restrict__ A,    // (B*S, Kp)
    const _Float16* __restrict__ W,    // (720, Kp)
    const float* __restrict__ bias,    // (720)
    _Float16* __restrict__ pre,        // (S, B, 720)
    int Kp) {
  const int lane = threadIdx.x & 31;
  const int wid  = threadIdx.x >> 5;
  const int tile = blockIdx.x * 8 + wid;               // 184320 tiles
  const int mt = tile / 45;
  const int nt = tile - mt * 45;
  const int m0 = mt << 4, n0 = nt << 4;
  const int col  = n0 + (lane & 15);
  const int hsel = lane >> 4;
  v8f acc;
  const float bv = bias[col];
#pragma unroll
  for (int v = 0; v < 8; ++v) acc[v] = bv;
  const _Float16* ap = A + (size_t)(m0 + (lane & 15)) * Kp + 8 * hsel;
  const _Float16* bp = W + (size_t)col * Kp + 8 * hsel;
  for (int kt = 0; kt < Kp; kt += 32) {
    v16h af = frag_ld(ap + kt);
    v16h bf = frag_ld(bp + kt);
    acc = __builtin_amdgcn_wmma_f32_16x16x32_f16(false, af, false, bf,
                                                 (short)0, acc, false, false);
  }
#pragma unroll
  for (int v = 0; v < 8; ++v) {
    int m = m0 + v + 8 * hsel;
    int b = m >> 10, t = m & (SS - 1);
    pre[((size_t)t * BB + b) * H4 + col] = (_Float16)acc[v];
  }
}

// ----------------------- persistent LSTM recurrence -------------------------
// Single workgroup (32 waves). LDS:
//   buf0/buf1: 64x720 f16 (92KB each)  -- pre[t] staged in by async loads,
//              then overwritten in place by the f16 gate values (each 16x16
//              tile reads pre and writes gates at identical positions, so the
//              aliasing is race-free across waves).
//   cbuf: 64x180 f32 (45KB), hbuf: 64x192 f16 (24KB).
// While step t computes, pre[t+1] streams into the other buffer via
// GLOBAL_LOAD_ASYNC_TO_LDS_B128; s_wait_asynccnt 0 lands a whole phase later.
#define PREB  (BB * H4 * 2)                 // 92160 bytes per pre buffer
#define SCAN_LDS (2 * PREB + BB * HH * 4 + BB * HPAD * 2)

__global__ void __launch_bounds__(1024, 1) lstm_scan_kernel(
    const _Float16* __restrict__ pre,   // (S, B, 720) f16
    const _Float16* __restrict__ whh,   // (720, 192) f16
    _Float16* __restrict__ hout) {      // (B, S, 192) f16
  extern __shared__ char smem[];
  _Float16* bufs[2];
  bufs[0] = (_Float16*)smem;
  bufs[1] = (_Float16*)(smem + PREB);
  float*    cbuf = (float*)(smem + 2 * PREB);
  _Float16* hbuf = (_Float16*)(smem + 2 * PREB + BB * HH * 4);
  const unsigned blds[2] = {lds_off32(bufs[0]), lds_off32(bufs[1])};

  const int tid  = threadIdx.x;
  const int lane = tid & 31, wid = tid >> 5;
  const int hsel = lane >> 4;

  for (int i = tid; i < BB * HH; i += 1024) cbuf[i] = 0.0f;
  for (int i = tid; i < BB * HPAD; i += 1024) hbuf[i] = (_Float16)0.0f;

  // stage pre[0] into buf0 (async DMA to LDS, tracked by ASYNCcnt)
  for (int idx = tid; idx < PREB / 16; idx += 1024) {
    asm volatile("global_load_async_to_lds_b128 %0, %1, off"
                 :: "v"(blds[0] + idx * 16),
                    "v"((unsigned long long)(uintptr_t)((const char*)pre + (size_t)idx * 16))
                 : "memory");
  }
  asm volatile("s_wait_asynccnt 0x0" ::: "memory");
  __syncthreads();

  for (int t = 0; t < SS; ++t) {
    _Float16* cur = bufs[t & 1];
    // kick off async staging of pre[t+1] into the other buffer
    if (t + 1 < SS) {
      const char* src = (const char*)pre + (size_t)(t + 1) * PREB;
      const unsigned dst = blds[(t + 1) & 1];
      for (int idx = tid; idx < PREB / 16; idx += 1024) {
        asm volatile("global_load_async_to_lds_b128 %0, %1, off"
                     :: "v"(dst + idx * 16),
                        "v"((unsigned long long)(uintptr_t)(src + (size_t)idx * 16))
                     : "memory");
      }
    }
    // gates = pre[t] + h @ Whh^T : 4 (mt) x 45 (nt) tiles round-robin on waves
    for (int tile = wid; tile < 180; tile += 32) {
      int mt = tile / 45, nt = tile - mt * 45;
      int m0 = mt << 4, n0 = nt << 4;
      int col = n0 + (lane & 15);
      v8f acc;
#pragma unroll
      for (int v = 0; v < 8; ++v)
        acc[v] = (float)cur[(m0 + v + 8 * hsel) * H4 + col];
      const _Float16* ap = hbuf + (m0 + (lane & 15)) * HPAD + 8 * hsel;
      const _Float16* bp = whh + (size_t)col * HPAD + 8 * hsel;
#pragma unroll
      for (int kt = 0; kt < HPAD; kt += 32) {
        v16h af = frag_ld(ap + kt);
        v16h bf = frag_ld(bp + kt);
        acc = __builtin_amdgcn_wmma_f32_16x16x32_f16(false, af, false, bf,
                                                     (short)0, acc, false, false);
      }
#pragma unroll
      for (int v = 0; v < 8; ++v)
        cur[(m0 + v + 8 * hsel) * H4 + col] = (_Float16)acc[v];   // gates, in place
    }
    __syncthreads();
    // elementwise c/h update (torch gate order i,f,g,o)
    for (int idx = tid; idx < BB * HH; idx += 1024) {
      int b = idx / HH, n = idx - b * HH;
      const _Float16* gb = cur + b * H4;
      float gi = sigmoidf_((float)gb[n]);
      float gf = sigmoidf_((float)gb[HH + n]);
      float gg = tanhf((float)gb[2 * HH + n]);
      float go = sigmoidf_((float)gb[3 * HH + n]);
      float c = gf * cbuf[idx] + gi * gg;
      cbuf[idx] = c;
      float h = go * tanhf(c);
      hbuf[b * HPAD + n] = (_Float16)h;
      hout[((size_t)b * SS + t) * HPAD + n] = (_Float16)h;
    }
    // zero the K-pad columns so next layer's GEMM reads zeros
    for (int idx = tid; idx < BB * (HPAD - HH); idx += 1024) {
      int b = idx / (HPAD - HH), n = HH + idx - b * (HPAD - HH);
      hout[((size_t)b * SS + t) * HPAD + n] = (_Float16)0.0f;
    }
    // next step reads the freshly staged buffer: drain our async loads, then
    // rendezvous so every wave's slice is visible.
    asm volatile("s_wait_asynccnt 0x0" ::: "memory");
    __syncthreads();
  }
}

// --------------------------- last-timestep gather ---------------------------
__global__ void __launch_bounds__(256) gather_last_kernel(
    const _Float16* __restrict__ h, const int* __restrict__ lens,
    _Float16* __restrict__ last) {
  int idx = blockIdx.x * 256 + threadIdx.x;   // (64, 192)
  if (idx >= BB * HPAD) return;
  int b = idx / HPAD, n = idx - b * HPAD;
  int t = lens[b] - 1;
  t = t < 0 ? 0 : (t > SS - 1 ? SS - 1 : t);
  last[idx] = (n < HH) ? h[((size_t)b * SS + t) * HPAD + n] : (_Float16)0.0f;
}

// ------------------------ post projection + LeakyReLU -----------------------
// (64,192) x (8192,192)^T -> (64,8192) f16, 4 x 512 tiles.
__global__ void __launch_bounds__(256) gemm_post_kernel(
    const _Float16* __restrict__ A, const _Float16* __restrict__ W,
    const float* __restrict__ bias, _Float16* __restrict__ out) {
  const int lane = threadIdx.x & 31;
  const int wid  = threadIdx.x >> 5;
  const int tile = blockIdx.x * 8 + wid;               // 2048 tiles
  const int mt = tile >> 9, nt = tile & 511;
  const int m0 = mt << 4, n0 = nt << 4;
  const int col = n0 + (lane & 15);
  const int hsel = lane >> 4;
  v8f acc;
  const float bv = bias[col];
#pragma unroll
  for (int v = 0; v < 8; ++v) acc[v] = bv;
  const _Float16* ap = A + (size_t)(m0 + (lane & 15)) * HPAD + 8 * hsel;
  const _Float16* bp = W + (size_t)col * HPAD + 8 * hsel;
#pragma unroll
  for (int kt = 0; kt < HPAD; kt += 32) {
    v16h af = frag_ld(ap + kt);
    v16h bf = frag_ld(bp + kt);
    acc = __builtin_amdgcn_wmma_f32_16x16x32_f16(false, af, false, bf,
                                                 (short)0, acc, false, false);
  }
#pragma unroll
  for (int v = 0; v < 8; ++v) {
    int m = m0 + v + 8 * hsel;
    float x = acc[v];
    x = x > 0.0f ? x : 0.01f * x;                      // LeakyReLU(0.01)
    out[(size_t)m * POST + col] = (_Float16)x;
  }
}

// ------------------------------ up projection -------------------------------
// (64,8192) x (304,8192)^T -> d_out (64,300) f32, 4 x 19 tiles (N padded 304).
__global__ void __launch_bounds__(256) gemm_up_kernel(
    const _Float16* __restrict__ A, const _Float16* __restrict__ W,
    const float* __restrict__ bias, float* __restrict__ out) {
  const int lane = threadIdx.x & 31;
  const int wid  = threadIdx.x >> 5;
  const int tile = blockIdx.x * 8 + wid;
  if (tile >= 4 * 19) return;
  const int mt = tile / 19, nt = tile - mt * 19;
  const int m0 = mt << 4, n0 = nt << 4;
  const int col = n0 + (lane & 15);
  const int hsel = lane >> 4;
  v8f acc;
  const float bv = (col < OUTN) ? bias[col] : 0.0f;
#pragma unroll
  for (int v = 0; v < 8; ++v) acc[v] = bv;
  const _Float16* ap = A + (size_t)(m0 + (lane & 15)) * POST + 8 * hsel;
  const _Float16* bp = W + (size_t)col * POST + 8 * hsel;
  for (int kt = 0; kt < POST; kt += 32) {
    // pull fragment lines ~1KB ahead of the load clauses (global_prefetch_b8)
    __builtin_prefetch(ap + kt + 512, 0, 3);
    __builtin_prefetch(bp + kt + 512, 0, 3);
    v16h af = frag_ld(ap + kt);
    v16h bf = frag_ld(bp + kt);
    acc = __builtin_amdgcn_wmma_f32_16x16x32_f16(false, af, false, bf,
                                                 (short)0, acc, false, false);
  }
  if (col < OUTN) {
#pragma unroll
    for (int v = 0; v < 8; ++v) {
      int m = m0 + v + 8 * hsel;
      out[(size_t)m * OUTN + col] = acc[v];
    }
  }
}

// ------------------------------- orchestration ------------------------------
extern "C" void kernel_launch(void* const* d_in, const int* in_sizes, int n_in,
                              void* d_out, int out_size, void* d_ws, size_t ws_size,
                              hipStream_t stream) {
  (void)in_sizes; (void)n_in; (void)out_size; (void)ws_size;
  const float* x        = (const float*)d_in[0];
  const int*   lens     = (const int*)d_in[1];
  const float* conv_w   = (const float*)d_in[2];
  const float* conv_b   = (const float*)d_in[3];
  const float* wih0     = (const float*)d_in[4];
  const float* wih_rest = (const float*)d_in[5];
  const float* whh      = (const float*)d_in[6];
  const float* bih      = (const float*)d_in[7];
  const float* bhh      = (const float*)d_in[8];
  const float* post_w   = (const float*)d_in[9];
  const float* post_b   = (const float*)d_in[10];
  const float* up_w     = (const float*)d_in[11];
  const float* up_b     = (const float*)d_in[12];
  float* out = (float*)d_out;

  // workspace carve
  size_t off = 0;
  auto carve = [&](size_t bytes) {
    void* p = (char*)d_ws + off;
    off += (bytes + 255) & ~(size_t)255;
    return p;
  };
  float*    c1    = (float*)carve((size_t)BB * SS * MEL * 4);
  float*    c2    = (float*)carve((size_t)BB * SS * MEL * 4);
  _Float16* A0    = (_Float16*)carve((size_t)BB * SS * K0P * 2);
  _Float16* preb  = (_Float16*)carve((size_t)SS * BB * H4 * 2);     // 94MB
  _Float16* hA    = (_Float16*)carve((size_t)BB * SS * HPAD * 2);
  _Float16* hB    = (_Float16*)carve((size_t)BB * SS * HPAD * 2);
  _Float16* wp[4];
  wp[0] = (_Float16*)carve((size_t)H4 * K0P * 2);
  for (int l = 1; l < 4; ++l) wp[l] = (_Float16*)carve((size_t)H4 * HPAD * 2);
  _Float16* whhp[4];
  for (int l = 0; l < 4; ++l) whhp[l] = (_Float16*)carve((size_t)H4 * HPAD * 2);
  float*    biasc = (float*)carve((size_t)4 * H4 * 4);
  _Float16* postp = (_Float16*)carve((size_t)POST * HPAD * 2);
  _Float16* upp   = (_Float16*)carve((size_t)304 * POST * 2);
  _Float16* last16 = (_Float16*)carve((size_t)BB * HPAD * 2);
  _Float16* act16  = (_Float16*)carve((size_t)BB * POST * 2);

  // 1) mel smooth residual blocks
  const int melN = BB * SS * MEL;
  mel_block_kernel<<<(melN + 255) / 256, 256, 0, stream>>>(x,  conv_w, conv_b, c1, 0);
  mel_block_kernel<<<(melN + 255) / 256, 256, 0, stream>>>(c1, conv_w, conv_b, c2, 1);
  mel_block_kernel<<<(melN + 255) / 256, 256, 0, stream>>>(c2, conv_w, conv_b, c1, 2);

  // 2) pack activations + weights to padded f16
  pack_a0_kernel<<<(BB * SS * K0P + 255) / 256, 256, 0, stream>>>(c1, A0);
  pack_w_kernel<<<(H4 * K0P + 255) / 256, 256, 0, stream>>>(wih0, wp[0], H4, H4, MEL, K0P);
  for (int l = 1; l < 4; ++l)
    pack_w_kernel<<<(H4 * HPAD + 255) / 256, 256, 0, stream>>>(
        wih_rest + (size_t)(l - 1) * H4 * HH, wp[l], H4, H4, HH, HPAD);
  for (int l = 0; l < 4; ++l)
    pack_w_kernel<<<(H4 * HPAD + 255) / 256, 256, 0, stream>>>(
        whh + (size_t)l * H4 * HH, whhp[l], H4, H4, HH, HPAD);
  pack_w_kernel<<<(POST * HPAD + 255) / 256, 256, 0, stream>>>(post_w, postp, POST, POST, HH, HPAD);
  pack_w_kernel<<<(304 * POST + 255) / 256, 256, 0, stream>>>(up_w, upp, 304, OUTN, POST, POST);
  bias_comb_kernel<<<(4 * H4 + 255) / 256, 256, 0, stream>>>(bih, bhh, biasc);

  // 3) LSTM layers: big input-proj GEMM, then persistent recurrence
  hipFuncSetAttribute((const void*)lstm_scan_kernel,
                      hipFuncAttributeMaxDynamicSharedMemorySize, SCAN_LDS);
  const int preBlocks = (4096 * 45) / 8;   // one wave per 16x16 tile
  const _Float16* Acur = A0;
  int KpCur = K0P;
  _Float16* houts[4] = {hA, hB, hA, hB};
  for (int l = 0; l < 4; ++l) {
    gemm_pre_kernel<<<preBlocks, 256, 0, stream>>>(Acur, wp[l], biasc + l * H4,
                                                   preb, KpCur);
    lstm_scan_kernel<<<1, 1024, SCAN_LDS, stream>>>(preb, whhp[l], houts[l]);
    Acur = houts[l];
    KpCur = HPAD;
  }

  // 4) gather last valid timestep, post (LeakyReLU) and up projections
  gather_last_kernel<<<(BB * HPAD + 255) / 256, 256, 0, stream>>>(houts[3], lens, last16);
  gemm_post_kernel<<<(4 * 512) / 8, 256, 0, stream>>>(last16, postp, post_b, act16);
  gemm_up_kernel<<<(4 * 19 + 7) / 8, 256, 0, stream>>>(act16, upp, up_b, out);
}
